// NeuralRadianceCache_59081570124136
// MI455X (gfx1250) — matrix operand order repaired and checked by reading, not compiled, not run on hardware
//
#include <hip/hip_runtime.h>
#include <hip/hip_bf16.h>
#include <stdint.h>

typedef __attribute__((ext_vector_type(16))) _Float16 v16h;
typedef __attribute__((ext_vector_type(8)))  _Float16 v8h;
typedef __attribute__((ext_vector_type(2)))  __fp16   v2fp;   // cvt_pkrtz return type
typedef __attribute__((ext_vector_type(8)))  float    v8f;
typedef __attribute__((ext_vector_type(4)))  int      v4i;

#define N_PTS    524288
#define N_LEVELS 16
#define LOG2_T   19
#define T_SIZE   (1u << LOG2_T)
#define ENC_PAD  64          // input features padded 36 -> 64 (2 WMMA k-steps)
#define NEURONS  128
#define TILE     128         // points per workgroup tile

// ---- workspace layout ----
// f16 weights, transposed [out][in]:
//   wIn^T  [128][64]     f16 :  16384 B @ 0
//   wHid^T [7][128][128] f16 : 229376 B @ 16384
//   wOut^T [16][128]     f16 :   4096 B @ 245760  (read direct from global)
#define W_IN_ELEMS   8192
#define W_HID_ELEMS  114688
#define W_OUT_ELEMS  2048
#define W_LDS_ELEMS  (W_IN_ELEMS + W_HID_ELEMS)          // 122880 staged in LDS
#define W_ALL_ELEMS  (W_LDS_ELEMS + W_OUT_ELEMS)         // 124928 in ws
#define WS_ENC_OFF   262144                              // enc f16 [N][64]

// tcnn level params: scale = 16*1.5^l - 1 ; res = ceil(scale)+1
__constant__ float kScale[N_LEVELS] = {
    15.f, 23.f, 35.f, 53.f, 80.f, 120.5f, 181.25f, 272.375f,
    409.0625f, 614.09375f, 921.640625f, 1382.9609375f,
    2074.94140625f, 3112.912109375f, 4669.8681640625f, 7005.30224609375f};
__constant__ uint32_t kRes[4] = {16u, 24u, 36u, 54u};    // dense levels 0..3

// ---------------- async global->LDS copy (CDNA5) ----------------
#if defined(__has_builtin)
#  if __has_builtin(__builtin_amdgcn_global_load_async_to_lds_b128)
#    define HAS_ASYNC_LDS 1
#  endif
#endif
#ifndef HAS_ASYNC_LDS
#  define HAS_ASYNC_LDS 0
#endif

__device__ __forceinline__ void cp_b128(void* lds_dst, const void* gsrc)
{
#if HAS_ASYNC_LDS
    __builtin_amdgcn_global_load_async_to_lds_b128(
        (v4i*)gsrc, (v4i*)lds_dst, 0, 0);
#else
    *(uint4*)lds_dst = *(const uint4*)gsrc;
#endif
}

__device__ __forceinline__ void wait_async()
{
#if HAS_ASYNC_LDS
#  if __has_builtin(__builtin_amdgcn_s_wait_asynccnt)
    __builtin_amdgcn_s_wait_asynccnt(0);
#  else
    asm volatile("s_wait_asynccnt 0x0" ::: "memory");
#  endif
#endif
}

// ------------------------------------------------------------------
// Kernel 0: f32 weights -> f16, transpose to [out][in], pad.
// ------------------------------------------------------------------
__global__ void nrc_prep_weights(const float* __restrict__ w_in,
                                 const float* __restrict__ w_hidden,
                                 const float* __restrict__ w_out,
                                 _Float16* __restrict__ wsW)
{
    int i = blockIdx.x * blockDim.x + threadIdx.x;
    if (i >= W_ALL_ELEMS) return;
    float v;
    if (i < W_IN_ELEMS) {                       // wIn^T [o][k], k padded to 64
        int o = i >> 6, k = i & 63;
        v = (k < 36) ? w_in[k * 128 + o] : 0.0f;
    } else if (i < W_LDS_ELEMS) {               // wHid^T [l][o][k]
        int j = i - W_IN_ELEMS;
        int l = j >> 14, r = j & 16383;
        int o = r >> 7, k = r & 127;
        v = w_hidden[l * 16384 + k * 128 + o];
    } else {                                    // wOut^T [o][k], o padded to 16
        int j = i - W_LDS_ELEMS;
        int o = j >> 7, k = j & 127;
        v = (o < 3) ? w_out[k * 3 + o] : 0.0f;
    }
    wsW[i] = (_Float16)v;
}

// ------------------------------------------------------------------
// Kernel 1: hash-grid encode -> f16 features [N][64] (zero padded).
// 64MB of tables live in L2; 8 gathers x 16 levels per point.
// ------------------------------------------------------------------
__global__ void nrc_encode(const float* __restrict__ x,
                           const float* __restrict__ tables,
                           _Float16* __restrict__ enc)
{
    int p = blockIdx.x * blockDim.x + threadIdx.x;
    if (p >= N_PTS) return;

    const float px = x[p * 16 + 0];
    const float py = x[p * 16 + 1];
    const float pz = x[p * 16 + 2];

    _Float16 o[ENC_PAD];

    #pragma unroll
    for (int l = 0; l < N_LEVELS; ++l) {
        const float s  = kScale[l];
        const float fx = px * s + 0.5f, fy = py * s + 0.5f, fz = pz * s + 0.5f;
        const float bx = floorf(fx),   by = floorf(fy),   bz = floorf(fz);
        const float rx = fx - bx,      ry = fy - by,      rz = fz - bz;
        const uint32_t ix = (uint32_t)bx, iy = (uint32_t)by, iz = (uint32_t)bz;
        const float2* tl = (const float2*)(tables) + (size_t)l * T_SIZE;

        float f0 = 0.f, f1 = 0.f;
        #pragma unroll
        for (int c = 0; c < 8; ++c) {
            const uint32_t dx = c & 1, dy = (c >> 1) & 1, dz = (c >> 2) & 1;
            const uint32_t cx = ix + dx, cy = iy + dy, cz = iz + dz;
            const float w = (dx ? rx : 1.f - rx) *
                            (dy ? ry : 1.f - ry) *
                            (dz ? rz : 1.f - rz);
            uint32_t idx;
            if (l < 4) {
                const uint32_t r = kRes[l];
                idx = (cx + cy * r + cz * r * r) & (T_SIZE - 1u);
            } else {
                idx = (cx * 1u ^ cy * 2654435761u ^ cz * 805459861u) & (T_SIZE - 1u);
            }
            const float2 g = tl[idx];
            f0 += w * g.x;
            f1 += w * g.y;
        }
        o[2 * l + 0] = (_Float16)f0;
        o[2 * l + 1] = (_Float16)f1;
    }
    #pragma unroll
    for (int d = 0; d < 4; ++d) o[32 + d] = (_Float16)x[p * 16 + 3 + d];
    #pragma unroll
    for (int d = 36; d < ENC_PAD; ++d) o[d] = (_Float16)0.f;

    uint4* dst = (uint4*)(enc + (size_t)p * ENC_PAD);
    const uint4* src = (const uint4*)o;
    #pragma unroll
    for (int i = 0; i < 8; ++i) dst[i] = src[i];
}

// ------------------------------------------------------------------
// Kernel 2: fully fused MLP, transposed-product formulation.
//   H_next^T[n][m] = sum_k Wt[n][k] * H^T[k][m]
//   A = Wt tiles (LDS, contiguous-K),  B = activations [point][feat],
//   D packs 8 contiguous neurons/lane -> ONE ds_store_b128 per tile.
// kt-outer / nt-inner loop keeps 8 independent WMMA accumulator
// chains in flight (no RAW between consecutive v_wmma issues).
// ------------------------------------------------------------------
__device__ __forceinline__ v16h ld_a(const _Float16* p)   // A frag: K {0..7},{16..23} per half
{
    union { v16h v; v8h h[2]; } u;
    u.h[0] = *(const v8h*)(p);
    u.h[1] = *(const v8h*)(p + 16);
    return u.v;
}
__device__ __forceinline__ v16h ld_b(const _Float16* p)   // B frag: K {0..7},{16..23} per half
{
    union { v16h v; v8h h[2]; } u;
    u.h[0] = *(const v8h*)(p);
    u.h[1] = *(const v8h*)(p + 16);
    return u.v;
}
__device__ __forceinline__ void st_relu_pack(_Float16* dst, v8f acc)
{
    union { v8h v; v2fp h[4]; } u;
    #pragma unroll
    for (int i = 0; i < 4; ++i)
        u.h[i] = __builtin_amdgcn_cvt_pkrtz(acc[2 * i], acc[2 * i + 1]);
    v8h z = {};
#if defined(__has_builtin) && __has_builtin(__builtin_elementwise_maximumnum)
    *(v8h*)dst = __builtin_elementwise_maximumnum(u.v, z);   // v_pk_max_num_f16
#else
    *(v8h*)dst = __builtin_elementwise_max(u.v, z);
#endif
}

__global__ __launch_bounds__(256, 1)
void nrc_mlp(const _Float16* __restrict__ wsW,
             const _Float16* __restrict__ enc,
             float* __restrict__ out)
{
    extern __shared__ _Float16 smem[];
    _Float16* sW   = smem;                    // wIn^T + wHid^T : 122880 f16
    _Float16* sEnc = smem + W_LDS_ELEMS;      // [128][64]      :   8192 f16
    _Float16* sA   = sEnc + TILE * ENC_PAD;   // [128][128]     :  16384 f16
    _Float16* sB   = sA + TILE * NEURONS;     // [128][128]     :  16384 f16
    // total = 163840 f16 = 327680 B = full 320KB WGP LDS

    const int tid  = threadIdx.x;
    const int lane = tid & 31;
    const int wave = tid >> 5;
    const int ln16 = lane & 15;   // A row (=neuron) / B col (=point)
    const int half = lane >> 4;
    const int m0   = wave * 16;   // this wave's 16 points
    const int NT   = N_PTS / TILE;
    const int step = gridDim.x;

    const _Float16* sWh = sW + W_IN_ELEMS;
    const _Float16* gWo = wsW + W_LDS_ELEMS;  // wOut^T [16][128], stays in global

    // ---- stage all resident weights into LDS (async) ----
    for (int i = tid; i < W_LDS_ELEMS / 8; i += 256)
        cp_b128((uint4*)sW + i, (const uint4*)wsW + i);

    // ---- prefetch first enc tile ----
    int tile = blockIdx.x;
    if (tile < NT) {
        const uint4* g = (const uint4*)(enc + (size_t)tile * TILE * ENC_PAD);
        #pragma unroll
        for (int i = 0; i < 4; ++i)
            cp_b128((uint4*)sEnc + tid + 256 * i, g + tid + 256 * i);
    }

    for (; tile < NT; tile += step) {
        wait_async();
        __syncthreads();   // weights (1st iter) + enc tile ready; act bufs free

        // ---- input layer: K=64 (padded), src sEnc (stride 64) -> sA ----
        {
            v8f acc[8] = {};
            #pragma unroll
            for (int kt = 0; kt < 2; ++kt) {
                const v16h b = ld_b(sEnc + (m0 + ln16) * ENC_PAD + kt * 32 + half * 8);
                #pragma unroll
                for (int nt = 0; nt < 8; ++nt)
                    acc[nt] = __builtin_amdgcn_wmma_f32_16x16x32_f16(
                        false, ld_a(sW + (nt * 16 + ln16) * ENC_PAD + kt * 32 + half * 8),
                        false, b, (short)0, acc[nt], false, false);
            }
            #pragma unroll
            for (int nt = 0; nt < 8; ++nt)
                st_relu_pack(sA + (m0 + ln16) * NEURONS + nt * 16 + half * 8, acc[nt]);
        }
        __syncthreads();   // everyone done reading sEnc

        // ---- overlap: async-prefetch next tile's enc during hidden layers ----
        {
            const int next = tile + step;
            if (next < NT) {
                const uint4* g = (const uint4*)(enc + (size_t)next * TILE * ENC_PAD);
                #pragma unroll
                for (int i = 0; i < 4; ++i)
                    cp_b128((uint4*)sEnc + tid + 256 * i, g + tid + 256 * i);
            }
        }

        // ---- 7 hidden layers: K=128, ping-pong sA<->sB ----
        _Float16* src = sA;
        _Float16* dst = sB;
        for (int L = 0; L < 7; ++L) {
            const _Float16* w = sWh + L * 16384;
            v8f acc[8] = {};
            #pragma unroll
            for (int kt = 0; kt < 4; ++kt) {
                const v16h b = ld_b(src + (m0 + ln16) * NEURONS + kt * 32 + half * 8);
                #pragma unroll
                for (int nt = 0; nt < 8; ++nt)
                    acc[nt] = __builtin_amdgcn_wmma_f32_16x16x32_f16(
                        false, ld_a(w + (nt * 16 + ln16) * NEURONS + kt * 32 + half * 8),
                        false, b, (short)0, acc[nt], false, false);
            }
            #pragma unroll
            for (int nt = 0; nt < 8; ++nt)
                st_relu_pack(dst + (m0 + ln16) * NEURONS + nt * 16 + half * 8, acc[nt]);
            __syncthreads();
            _Float16* t = src; src = dst; dst = t;
        }
        // after 7 layers, final activations sit in sB (src == sB)

        // ---- output layer: A = wOut^T direct from global, 1 N-tile ----
        {
            v8f acc = {};
            #pragma unroll
            for (int kt = 0; kt < 4; ++kt)
                acc = __builtin_amdgcn_wmma_f32_16x16x32_f16(
                    false, ld_a(gWo + ln16 * NEURONS + kt * 32 + half * 8),
                    false, ld_b(src + (m0 + ln16) * NEURONS + kt * 32 + half * 8),
                    (short)0, acc, false, false);
            if (half == 0) {                       // lanes holding neurons 0..7
                const int m = tile * TILE + m0 + ln16;
                out[m * 3 + 0] = acc[0];
                out[m * 3 + 1] = acc[1];
                out[m * 3 + 2] = acc[2];
            }
        }
        __syncthreads();   // protect act buffers before next tile
    }
}

// ------------------------------------------------------------------
extern "C" void kernel_launch(void* const* d_in, const int* in_sizes, int n_in,
                              void* d_out, int out_size, void* d_ws, size_t ws_size,
                              hipStream_t stream)
{
    const float* x        = (const float*)d_in[0];   // [N,16]
    const float* tables   = (const float*)d_in[1];   // [16,T,2]
    const float* w_in     = (const float*)d_in[2];   // [36,128]
    const float* w_hidden = (const float*)d_in[3];   // [7,128,128]
    const float* w_out    = (const float*)d_in[4];   // [128,3]
    float* out            = (float*)d_out;           // [N,3]

    _Float16* wsW = (_Float16*)d_ws;
    _Float16* enc = (_Float16*)((char*)d_ws + WS_ENC_OFF);

    nrc_prep_weights<<<(W_ALL_ELEMS + 255) / 256, 256, 0, stream>>>(
        w_in, w_hidden, w_out, wsW);

    nrc_encode<<<N_PTS / 256, 256, 0, stream>>>(x, tables, enc);

    const size_t lds_bytes =
        (size_t)(W_LDS_ELEMS + TILE * ENC_PAD + 2 * TILE * NEURONS) *
        sizeof(_Float16);   // 327680 B = 320KB
    nrc_mlp<<<1024, 256, lds_bytes, stream>>>(wsW, enc, out);
}
